// BatchMMDLoss_55843164782710
// MI455X (gfx1250) — matrix operand order repaired and checked by reading, not compiled, not run on hardware
//
#include <hip/hip_runtime.h>
#include <hip/hip_bf16.h>

typedef __attribute__((ext_vector_type(16))) __bf16 v16bf;
typedef __attribute__((ext_vector_type(8)))  float  v8f;

union Frag {
    v16bf v;
    float4 h[2];   // two 16-byte chunks = 8 bf16 each
};

#define NBATCH   32
#define NROWS    2048           // 2*1024
#define DFEAT    128
#define NBLK     16             // 2048/128 row-blocks
#define NPAIRS   136            // upper-triangular 16x17/2
#define QUAD     1024

// ---------------------------------------------------------------------------
// Kernel A: per-batch row squared norms + analytic bandwidth.
// sum(D2) = 2n*sum_i||z_i||^2 - 2*||sum_i z_i||^2   (exact, no Gram needed)
// ---------------------------------------------------------------------------
__global__ __launch_bounds__(256) void mmd_stats(const float* __restrict__ X,
                                                 const float* __restrict__ Y,
                                                 float* __restrict__ sq,
                                                 float* __restrict__ negInvBw) {
    const int b = blockIdx.x;
    const int t = threadIdx.x;
    const float* xb = X + (size_t)b * QUAD * DFEAT;
    const float* yb = Y + (size_t)b * QUAD * DFEAT;

    __shared__ float red[256];
    __shared__ float sfeat[DFEAT];

    // row squared norms + partial sum of them
    float s1 = 0.f;
    for (int r = t; r < NROWS; r += 256) {
        const float* row = (r < QUAD) ? (xb + (size_t)r * DFEAT)
                                      : (yb + (size_t)(r - QUAD) * DFEAT);
        const float4* r4 = (const float4*)row;
        float acc = 0.f;
        for (int k = 0; k < DFEAT / 4; ++k) {
            float4 v = r4[k];
            acc += v.x * v.x + v.y * v.y + v.z * v.z + v.w * v.w;
        }
        sq[(size_t)b * NROWS + r] = acc;
        s1 += acc;
    }
    red[t] = s1;
    __syncthreads();
    for (int s = 128; s > 0; s >>= 1) {
        if (t < s) red[t] += red[t + s];
        __syncthreads();
    }
    float S1 = red[0];

    // per-feature column sums: threads 0..127 do X half, 128..255 do Y half
    const int d = t & 127;
    const float* half = (t < 128) ? xb : yb;
    float fs = 0.f;
    for (int r = 0; r < QUAD; ++r) fs += half[(size_t)r * DFEAT + d];
    __syncthreads();                 // also protects red[0] read above
    if (t < 128) sfeat[d] = fs;
    __syncthreads();
    if (t >= 128) sfeat[d] += fs;
    __syncthreads();

    float s2p = (t < 128) ? sfeat[t] * sfeat[t] : 0.f;
    red[t] = s2p;
    __syncthreads();
    for (int s = 128; s > 0; s >>= 1) {
        if (t < s) red[t] += red[t + s];
        __syncthreads();
    }
    if (t == 0) {
        const float n = (float)NROWS;
        float sumD2 = 2.f * n * S1 - 2.f * red[0];
        float bw = sumD2 / (n * n - n);
        negInvBw[b] = -1.f / bw;
    }
}

// ---------------------------------------------------------------------------
// Kernel B: fused Gram (bf16x3 WMMA) -> D2 -> exp -> weighted partial sum.
// One WG = 128x128 output block (bi<=bj); 8 waves, each a 32x64 tile.
// Dynamic LDS: Ahi|Alo|Bhi|Blo, each 128x128 bf16 (32KB) = 128KB.
// ---------------------------------------------------------------------------
__global__ __launch_bounds__(256) void mmd_tiles(const float* __restrict__ X,
                                                 const float* __restrict__ Y,
                                                 const float* __restrict__ sq,
                                                 const float* __restrict__ negInvBw,
                                                 float* __restrict__ partials) {
    extern __shared__ char smem[];
    __bf16* Ahi = (__bf16*)smem;
    __bf16* Alo = Ahi + 128 * 128;
    __bf16* Bhi = Alo + 128 * 128;
    __bf16* Blo = Bhi + 128 * 128;
    __shared__ float red[256];

    const int b = blockIdx.y;
    // map linear pair index -> (bi, bj) with bi <= bj
    int bi = 0, rem = blockIdx.x;
    while (rem >= NBLK - bi) { rem -= NBLK - bi; ++bi; }
    const int bj = bi + rem;

    // 128-row panels never straddle the X/Y boundary (1024/128 == 8)
    const float* srcA = (bi < 8)
        ? (X + ((size_t)b * QUAD + (size_t)bi * 128) * DFEAT)
        : (Y + ((size_t)b * QUAD + (size_t)(bi - 8) * 128) * DFEAT);
    const float* srcB = (bj < 8)
        ? (X + ((size_t)b * QUAD + (size_t)bj * 128) * DFEAT)
        : (Y + ((size_t)b * QUAD + (size_t)(bj - 8) * 128) * DFEAT);

    // stage panels to LDS as bf16 hi/lo split (fp32 ~= hi + lo)
    for (int idx = threadIdx.x * 4; idx < 128 * DFEAT; idx += 256 * 4) {
        float4 va = *(const float4*)(srcA + idx);
        float4 vb = *(const float4*)(srcB + idx);
        float a[4] = {va.x, va.y, va.z, va.w};
        float c[4] = {vb.x, vb.y, vb.z, vb.w};
        #pragma unroll
        for (int e = 0; e < 4; ++e) {
            __bf16 ah = (__bf16)a[e];
            Ahi[idx + e] = ah;
            Alo[idx + e] = (__bf16)(a[e] - (float)ah);
            __bf16 bh = (__bf16)c[e];
            Bhi[idx + e] = bh;
            Blo[idx + e] = (__bf16)(c[e] - (float)bh);
        }
    }
    __syncthreads();

    const int wave   = threadIdx.x >> 5;
    const int lane   = threadIdx.x & 31;
    const int lhalf  = lane >> 4;      // 0: lanes 0-15, 1: lanes 16-31
    const int lmod   = lane & 15;
    const int rowbase = (wave & 3) * 32;   // wave's rows within 128
    const int colbase = (wave >> 2) * 64;  // wave's cols within 128

    v8f acc[2][4] = {};

    #pragma unroll
    for (int ks = 0; ks < 4; ++ks) {       // K = 128 in steps of 32
        const int kb = ks * 32;
        Frag ahi[2], alo[2];
        #pragma unroll
        for (int rt = 0; rt < 2; ++rt) {
            const int row = rowbase + rt * 16 + lmod;
            const int ko  = kb + lhalf * 8;             // ISA A-frag K offset
            const float4* pH = (const float4*)(Ahi + row * DFEAT + ko);
            const float4* pL = (const float4*)(Alo + row * DFEAT + ko);
            ahi[rt].h[0] = pH[0]; ahi[rt].h[1] = pH[2]; // +16 bf16 = +2 chunks
            alo[rt].h[0] = pL[0]; alo[rt].h[1] = pL[2];
        }
        Frag bhi[4], blo[4];
        #pragma unroll
        for (int ct = 0; ct < 4; ++ct) {
            const int row = colbase + ct * 16 + lmod;   // column of Z = row of Z
            const int ko  = kb + lhalf * 8;             // ISA B-frag K offset
            const float4* pH = (const float4*)(Bhi + row * DFEAT + ko);
            const float4* pL = (const float4*)(Blo + row * DFEAT + ko);
            bhi[ct].h[0] = pH[0]; bhi[ct].h[1] = pH[2];
            blo[ct].h[0] = pL[0]; blo[ct].h[1] = pL[2];
        }
        #pragma unroll
        for (int rt = 0; rt < 2; ++rt)
            #pragma unroll
            for (int ct = 0; ct < 4; ++ct) {
                acc[rt][ct] = __builtin_amdgcn_wmma_f32_16x16x32_bf16(
                    false, ahi[rt].v, false, bhi[ct].v, (short)0, acc[rt][ct], false, false);
                acc[rt][ct] = __builtin_amdgcn_wmma_f32_16x16x32_bf16(
                    false, ahi[rt].v, false, blo[ct].v, (short)0, acc[rt][ct], false, false);
                acc[rt][ct] = __builtin_amdgcn_wmma_f32_16x16x32_bf16(
                    false, alo[rt].v, false, bhi[ct].v, (short)0, acc[rt][ct], false, false);
            }
    }

    // epilogue: D2 = sq_i + sq_j - 2G ; exp ; quadrant weights (+1 same, -1 cross)
    const float* sqb = sq + (size_t)b * NROWS;
    const float nib = negInvBw[b];
    const float blockW = ((bi == bj) ? 1.f : 2.f) / (1048576.f);  // /1024^2, x2 off-diag
    float lsum = 0.f;
    #pragma unroll
    for (int rt = 0; rt < 2; ++rt) {
        const int ibase = bi * 128 + rowbase + rt * 16 + lhalf * 8;  // C layout: M=r+8*lhalf
        #pragma unroll
        for (int ct = 0; ct < 4; ++ct) {
            const int j = bj * 128 + colbase + ct * 16 + lmod;       // C layout: N=lane%16
            const float sj = sqb[j];
            const bool jq = (j < QUAD);
            #pragma unroll
            for (int r = 0; r < 8; ++r) {
                const int i = ibase + r;
                float d2 = sqb[i] + sj - 2.f * acc[rt][ct][r];
                d2 = fmaxf(d2, 0.f);
                const float e = __expf(d2 * nib);
                const bool iq = (i < QUAD);
                lsum += ((iq == jq) ? blockW : -blockW) * e;
            }
        }
    }

    red[threadIdx.x] = lsum;
    __syncthreads();
    for (int s = 128; s > 0; s >>= 1) {
        if (threadIdx.x < s) red[threadIdx.x] += red[threadIdx.x + s];
        __syncthreads();
    }
    if (threadIdx.x == 0)
        partials[(size_t)blockIdx.y * NPAIRS + blockIdx.x] = red[0];
}

// ---------------------------------------------------------------------------
// Kernel C: deterministic reduction of all per-WG partials -> scalar output.
// ---------------------------------------------------------------------------
__global__ __launch_bounds__(256) void mmd_reduce(const float* __restrict__ partials,
                                                  int n, float* __restrict__ out) {
    __shared__ float red[256];
    const int t = threadIdx.x;
    float s = 0.f;
    for (int i = t; i < n; i += 256) s += partials[i];
    red[t] = s;
    __syncthreads();
    for (int k = 128; k > 0; k >>= 1) {
        if (t < k) red[t] += red[t + k];
        __syncthreads();
    }
    if (t == 0) out[0] = red[0];
}

extern "C" void kernel_launch(void* const* d_in, const int* in_sizes, int n_in,
                              void* d_out, int out_size, void* d_ws, size_t ws_size,
                              hipStream_t stream) {
    const float* X = (const float*)d_in[0];   // [32,1024,128] fp32
    const float* Y = (const float*)d_in[1];   // [32,1024,128] fp32

    float* ws       = (float*)d_ws;
    float* sq       = ws;                       // 32*2048 floats
    float* nib      = ws + NBATCH * NROWS;      // 32 floats
    float* partials = nib + NBATCH;             // 136*32 floats

    mmd_stats<<<NBATCH, 256, 0, stream>>>(X, Y, sq, nib);

    dim3 grid(NPAIRS, NBATCH);
    size_t ldsBytes = (size_t)4 * 128 * 128 * sizeof(__bf16);  // 128 KB
    mmd_tiles<<<grid, 256, ldsBytes, stream>>>(X, Y, sq, nib, partials);

    mmd_reduce<<<1, 256, 0, stream>>>(partials, NPAIRS * NBATCH, (float*)d_out);
}
